// IPAGNNLayer_70995809403266
// MI455X (gfx1250) — matrix operand order repaired and checked by reading, not compiled, not run on hardware
//
#include <hip/hip_runtime.h>
#include <hip/hip_fp16.h>
#include <math.h>
#include <stdint.h>

#define BATCH 32
#define NNODE 2048
#define HDIM  128
#define GDIM  512   // 4H
#define KDIM  256   // 2H (fused [x,h])
#define MROWS (BATCH*NNODE)
#define OUTC  (GDIM+1) // 513
#define KSTEPS 8       // KDIM/32
#define BSTEP  16384   // halves per k-step of swizzled B: 4q * 8hblk * 32lane * 16e

typedef __attribute__((ext_vector_type(16))) _Float16 v16h;
typedef __attribute__((ext_vector_type(8)))  float    v8f;

__device__ __forceinline__ float sigmoidf_(float x) { return 1.0f / (1.0f + __expf(-x)); }

// ---------------------------------------------------------------------------
// Convert [Wi;Wh] (256x512 f32) -> f16 weights pre-swizzled into WMMA
// B-fragment order:  off = (((kstep*4 + q)*8 + hblk)*32 + lane)*16 + e
// where  k = kstep*32 + e + 16*(lane>>4),  g = q*128 + hblk*16 + (lane&15).
// A lane's whole fragment is then 32 contiguous bytes.
// ---------------------------------------------------------------------------
__global__ __launch_bounds__(256)
void prep_weights(const float* __restrict__ Wi0, const float* __restrict__ Wh0,
                  const float* __restrict__ Wi1, const float* __restrict__ Wh1,
                  _Float16* __restrict__ Wsw0, _Float16* __restrict__ Wsw1) {
    int id = blockIdx.x * blockDim.x + threadIdx.x;   // over (k,g)
    if (id >= KDIM * GDIM) return;
    int k = id / GDIM, g = id % GDIM;
    int kstep = k >> 5, kl = k & 31;
    int hiB = kl >> 4, e = kl & 15;
    int q = g >> 7, gl = g & 127;
    int hblk = gl >> 4, n = gl & 15;
    size_t dst = (size_t)(((kstep * 4 + q) * 8 + hblk) * 32 + (n + 16 * hiB)) * 16 + e;
    float a = (k < HDIM) ? Wi0[k * GDIM + g] : Wh0[(k - HDIM) * GDIM + g];
    float b = (k < HDIM) ? Wi1[k * GDIM + g] : Wh1[(k - HDIM) * GDIM + g];
    Wsw0[dst] = (_Float16)a;
    Wsw1[dst] = (_Float16)b;
}

// ---------------------------------------------------------------------------
// Fused LSTM layer via v_wmma_f32_16x16x32_f16.
// Block = 8 waves, 64 rows (4 row-tiles).  Wave w owns h-block w; its four
// gate quadrants (i,f,g,o) are accumulated so the LSTM nonlinearity fuses
// into the GEMM epilogue.  A tiles: cooperative global->reg->LDS convert in
// fragment swizzle (ds_load_b128 feeds).  B tiles: async global->LDS copies
// (ASYNCcnt) of the pre-swizzled f16 weights, double-buffered per k-step.
// ---------------------------------------------------------------------------
__global__ __launch_bounds__(256)
void lstm_wmma(const float* __restrict__ A0,      // (M,H): K rows 0..127
               const float* __restrict__ A1,      // (M,H): K rows 128..255
               const _Float16* __restrict__ Bsw,  // swizzled (256x512) f16
               const float* __restrict__ bias,    // (512,)
               const float* __restrict__ Cprev,   // (M,H)
               float* __restrict__ Cout,
               float* __restrict__ Hout) {
    __shared__ __align__(16) _Float16 As[2][4][512];    // [buf][rowtile][lane*16+e]
    __shared__ __align__(16) _Float16 Bs[2][BSTEP];     // [buf][(q*8+hblk)*512 + lane*16+e]

    const int tx   = threadIdx.x;          // lane 0..31
    const int wave = threadIdx.y;          // 0..7 -> h-block
    const int tid  = wave * 32 + tx;
    const int row0 = blockIdx.x * 64;
    const int hi   = tx >> 4;
    const int lan  = tx & 15;

    v8f acc[4][4] = {};                    // [rowtile][gate]

    auto loadA = [&](int kstep, int buf) {
        // 64 rows x 32 K f32 -> f16 fragment swizzle; 8 contiguous f32 per thread.
        int row    = tid >> 2;             // 0..63
        int kgroup = tid & 3;              // 8-wide K chunk
        const float* src;
        int kb;
        if (kstep < 4) { src = A0; kb = kstep * 32; }
        else           { src = A1; kb = (kstep - 4) * 32; }
        const float* p = src + (size_t)(row0 + row) * HDIM + kb + kgroup * 8;
        int hiA   = kgroup >> 1;           // klocal>=16
        int ebase = (kgroup & 1) * 8;      // klocal>=8(&&<16) or >=24
        union { _Float16 h[8]; uint4 u; } t;
#pragma unroll
        for (int j = 0; j < 8; ++j) t.h[j] = (_Float16)p[j];
        *(uint4*)&As[buf][row >> 4][((row & 15) + 16 * hiA) * 16 + ebase] = t.u;
    };

    auto asyncB = [&](int kstep, int buf) {
        // 32KB of pre-swizzled B, 2048 x 16B chunks, 8 per thread.
        const _Float16* g = Bsw + (size_t)kstep * BSTEP;
        _Float16* l = &Bs[buf][0];
#pragma unroll
        for (int i = 0; i < 8; ++i) {
            unsigned c = (unsigned)(i * 256 + tid);
            unsigned           lo = (unsigned)(uintptr_t)(l + (size_t)c * 8);
            unsigned long long ga = (unsigned long long)(uintptr_t)(g + (size_t)c * 8);
            asm volatile("global_load_async_to_lds_b128 %0, %1, off"
                         :: "v"(lo), "v"(ga) : "memory");
        }
    };

    auto compute = [&](int buf) {
        v16h bfrag[4];
#pragma unroll
        for (int q = 0; q < 4; ++q)
            bfrag[q] = *(const v16h*)&Bs[buf][(size_t)((q * 8 + wave) * 32 + tx) * 16];
#pragma unroll
        for (int r = 0; r < 4; ++r) {
            v16h afrag = *(const v16h*)&As[buf][r][tx * 16];
#pragma unroll
            for (int q = 0; q < 4; ++q)
                acc[r][q] = __builtin_amdgcn_wmma_f32_16x16x32_f16(
                    false, afrag, false, bfrag[q], (short)0, acc[r][q], false, false);
        }
    };

    // prologue: fill buffer 0
    loadA(0, 0);
    asyncB(0, 0);
    asm volatile("s_wait_asynccnt 0x0" ::: "memory");
    __syncthreads();

    for (int k = 0; k < KSTEPS; ++k) {
        int cur = k & 1, nxt = cur ^ 1;
        if (k < KSTEPS - 1) {
            asyncB(k + 1, nxt);
            loadA(k + 1, nxt);
        }
        compute(cur);
        asm volatile("s_wait_asynccnt 0x0" ::: "memory");
        __syncthreads();
    }

    // Epilogue: D layout -> lane column n=lan, row m = v + 8*hi.
    const int h  = wave * 16 + lan;
    const float bi = bias[h];
    const float bf = bias[HDIM + h];
    const float bg = bias[2 * HDIM + h];
    const float bo = bias[3 * HDIM + h];
#pragma unroll
    for (int r = 0; r < 4; ++r) {
#pragma unroll
        for (int v = 0; v < 8; ++v) {
            int m = v + 8 * hi;
            size_t idx = (size_t)(row0 + r * 16 + m) * HDIM + h;
            float gi = acc[r][0][v] + bi;
            float gf = acc[r][1][v] + bf;
            float gg = acc[r][2][v] + bg;
            float go = acc[r][3][v] + bo;
            float cp = Cprev[idx];
            float cn = sigmoidf_(gf) * cp + sigmoidf_(gi) * tanhf(gg);
            float hn = sigmoidf_(go) * tanhf(cn);
            Cout[idx] = cn;
            Hout[idx] = hn;
        }
    }
}

// ---------------------------------------------------------------------------
// Restore old state at exit/raise nodes (mask_h).
// ---------------------------------------------------------------------------
__global__ __launch_bounds__(128)
void mask_nodes_k(float* c0n, float* h0n, float* c1n, float* h1n,
                  const float* oc0, const float* oh0,
                  const float* oc1, const float* oh1,
                  const int* exit_idx, const int* raise_idx) {
    int b = blockIdx.x >> 1;
    int node = (blockIdx.x & 1) ? raise_idx[b] : exit_idx[b];
    size_t base = ((size_t)b * NNODE + node) * HDIM + threadIdx.x;
    c0n[base] = oc0[base];
    h0n[base] = oh0[base];
    c1n[base] = oc1[base];
    h1n[base] = oh1[base];
}

// ---------------------------------------------------------------------------
// raise/branch heads: one wave per node; 512-wide dot products reduced with
// wave32 shuffles; softmax-2; scatter ip mass to ip_new with f32 atomics.
// ---------------------------------------------------------------------------
__global__ __launch_bounds__(256)
void decisions_k(const float* __restrict__ c0n, const float* __restrict__ h0n,
                 const float* __restrict__ c1n, const float* __restrict__ h1n,
                 const float* __restrict__ Wr, const float* __restrict__ br,
                 const float* __restrict__ Wb, const float* __restrict__ bb,
                 const float* __restrict__ ip,
                 const int* __restrict__ exit_idx, const int* __restrict__ raise_node_idx,
                 const int* __restrict__ true_idx, const int* __restrict__ false_idx,
                 const int* __restrict__ raise_idx,
                 float* wr, float* wt, float* wf, float* ip_new) {
    int lane = threadIdx.x & 31;
    int wv   = threadIdx.x >> 5;
    int node = blockIdx.x * 8 + wv;          // 0..MROWS-1 (exact)
    int b = node / NNODE, n = node % NNODE;

    const float* bufs[4] = {c0n, h0n, c1n, h1n};
    float r0 = 0.f, r1 = 0.f, q0 = 0.f, q1 = 0.f;
#pragma unroll
    for (int s = 0; s < 4; ++s) {
        const float* p = bufs[s] + (size_t)node * HDIM;
        for (int j = lane; j < HDIM; j += 32) {
            float v = p[j];
            int idx = s * HDIM + j;
            r0 += v * Wr[idx * 2 + 0];
            r1 += v * Wr[idx * 2 + 1];
            q0 += v * Wb[idx * 2 + 0];
            q1 += v * Wb[idx * 2 + 1];
        }
    }
    for (int off = 16; off; off >>= 1) {
        r0 += __shfl_xor(r0, off, 32);
        r1 += __shfl_xor(r1, off, 32);
        q0 += __shfl_xor(q0, off, 32);
        q1 += __shfl_xor(q1, off, 32);
    }
    if (lane == 0) {
        r0 += br[0]; r1 += br[1];
        q0 += bb[0]; q1 += bb[1];
        float mr = fmaxf(r0, r1);
        float e0 = __expf(r0 - mr), e1 = __expf(r1 - mr);
        float pr = e0 / (e0 + e1), pn = e1 / (e0 + e1);
        if (n == exit_idx[b] || n == raise_node_idx[b]) { pr = 0.f; pn = 1.f; }
        float mb = fmaxf(q0, q1);
        float f0 = __expf(q0 - mb), f1 = __expf(q1 - mb);
        float pt = f0 / (f0 + f1), pf = f1 / (f0 + f1);
        float ipv = ip[node];
        float wrv = pr * ipv, wtv = pn * pt * ipv, wfv = pn * pf * ipv;
        wr[node] = wrv; wt[node] = wtv; wf[node] = wfv;
        atomicAdd(&ip_new[b * NNODE + raise_idx[node]], wrv);
        atomicAdd(&ip_new[b * NNODE + true_idx[node]],  wtv);
        atomicAdd(&ip_new[b * NNODE + false_idx[node]], wfv);
    }
}

// ---------------------------------------------------------------------------
// Weighted segment-sum of hidden states directly into packed output (zeroed).
// ---------------------------------------------------------------------------
__global__ __launch_bounds__(256)
void aggregate_k(const float* __restrict__ c0n, const float* __restrict__ h0n,
                 const float* __restrict__ c1n, const float* __restrict__ h1n,
                 const float* __restrict__ wr, const float* __restrict__ wt,
                 const float* __restrict__ wf,
                 const int* __restrict__ true_idx, const int* __restrict__ false_idx,
                 const int* __restrict__ raise_idx,
                 float* __restrict__ out) {
    size_t id = (size_t)blockIdx.x * blockDim.x + threadIdx.x;
    if (id >= (size_t)MROWS * HDIM) return;
    int node = (int)(id / HDIM), h = (int)(id % HDIM);
    int b = node / NNODE;
    float wrv = wr[node], wtv = wt[node], wfv = wf[node];
    size_t rb = ((size_t)b * NNODE + raise_idx[node]) * OUTC;
    size_t tb = ((size_t)b * NNODE + true_idx[node])  * OUTC;
    size_t fb = ((size_t)b * NNODE + false_idx[node]) * OUTC;
    const float* bufs[4] = {c0n, h0n, c1n, h1n};
#pragma unroll
    for (int s = 0; s < 4; ++s) {
        float v = bufs[s][(size_t)node * HDIM + h];
        int c = s * HDIM + h;
        atomicAdd(&out[rb + c], v * wrv);
        atomicAdd(&out[tb + c], v * wtv);
        atomicAdd(&out[fb + c], v * wfv);
    }
}

// ---------------------------------------------------------------------------
// Divide by denom, keep_old_if_done, pack instruction pointer.
// ---------------------------------------------------------------------------
__global__ __launch_bounds__(256)
void finalize_k(float* __restrict__ out, const float* __restrict__ ip_new,
                const float* __restrict__ ip_old,
                const float* __restrict__ oc0, const float* __restrict__ oh0,
                const float* __restrict__ oc1, const float* __restrict__ oh1,
                const int* __restrict__ step_limits, const int* __restrict__ current_step) {
    size_t id = (size_t)blockIdx.x * blockDim.x + threadIdx.x;
    if (id >= (size_t)MROWS * OUTC) return;
    int node = (int)(id / OUTC), c = (int)(id % OUTC);
    int b = node / NNODE;
    bool nd = current_step[0] < step_limits[b];
    if (c < GDIM) {
        if (nd) {
            out[id] = out[id] / (ip_new[node] + 1e-7f);
        } else {
            const float* olds[4] = {oc0, oh0, oc1, oh1};
            out[id] = olds[c / HDIM][(size_t)node * HDIM + (c % HDIM)];
        }
    } else {
        out[id] = nd ? ip_new[node] : ip_old[node];
    }
}

// ---------------------------------------------------------------------------
extern "C" void kernel_launch(void* const* d_in, const int* in_sizes, int n_in,
                              void* d_out, int out_size, void* d_ws, size_t ws_size,
                              hipStream_t stream) {
    const float* node_emb   = (const float*)d_in[0];
    const float* hs_c0      = (const float*)d_in[1];
    const float* hs_h0      = (const float*)d_in[2];
    const float* hs_c1      = (const float*)d_in[3];
    const float* hs_h1      = (const float*)d_in[4];
    const float* ip_old     = (const float*)d_in[5];
    const float* Wi0        = (const float*)d_in[6];
    const float* Wh0        = (const float*)d_in[7];
    const float* b0         = (const float*)d_in[8];
    const float* Wi1        = (const float*)d_in[9];
    const float* Wh1        = (const float*)d_in[10];
    const float* b1         = (const float*)d_in[11];
    const float* W_raise    = (const float*)d_in[12];
    const float* b_raise    = (const float*)d_in[13];
    const float* W_branch   = (const float*)d_in[14];
    const float* b_branch   = (const float*)d_in[15];
    const int*   true_idx   = (const int*)d_in[16];
    const int*   false_idx  = (const int*)d_in[17];
    const int*   raise_idx  = (const int*)d_in[18];
    const int*   exit_nidx  = (const int*)d_in[19];
    const int*   raise_nidx = (const int*)d_in[20];
    const int*   step_lim   = (const int*)d_in[21];
    const int*   cur_step   = (const int*)d_in[22];

    float* out = (float*)d_out;

    // --- workspace carve ---
    char* ws = (char*)d_ws;
    size_t off = 0;
    _Float16* Wsw0 = (_Float16*)(ws + off); off += (size_t)KDIM * GDIM * sizeof(_Float16);
    _Float16* Wsw1 = (_Float16*)(ws + off); off += (size_t)KDIM * GDIM * sizeof(_Float16);
    float* c0n = (float*)(ws + off); off += (size_t)MROWS * HDIM * sizeof(float);
    float* h0n = (float*)(ws + off); off += (size_t)MROWS * HDIM * sizeof(float);
    float* c1n = (float*)(ws + off); off += (size_t)MROWS * HDIM * sizeof(float);
    float* h1n = (float*)(ws + off); off += (size_t)MROWS * HDIM * sizeof(float);
    float* wr  = (float*)(ws + off); off += (size_t)MROWS * sizeof(float);
    float* wt  = (float*)(ws + off); off += (size_t)MROWS * sizeof(float);
    float* wf  = (float*)(ws + off); off += (size_t)MROWS * sizeof(float);
    float* ipn = (float*)(ws + off); off += (size_t)MROWS * sizeof(float);
    (void)ws_size; (void)in_sizes; (void)n_in;

    // Zero the atomic-accumulation targets (graph-capture-safe async memsets).
    hipMemsetAsync(d_out, 0, (size_t)out_size * sizeof(float), stream);
    hipMemsetAsync(ipn, 0, (size_t)MROWS * sizeof(float), stream);

    // 1) weights -> f16, pre-swizzled into B-fragment order
    prep_weights<<<(KDIM * GDIM + 255) / 256, 256, 0, stream>>>(Wi0, Wh0, Wi1, Wh1, Wsw0, Wsw1);

    // 2) two fused LSTM layers via WMMA (64 rows per block)
    dim3 blk(32, 8);
    lstm_wmma<<<MROWS / 64, blk, 0, stream>>>(node_emb, hs_h0, Wsw0, b0, hs_c0, c0n, h0n);
    lstm_wmma<<<MROWS / 64, blk, 0, stream>>>(h0n,      hs_h1, Wsw1, b1, hs_c1, c1n, h1n);

    // 3) mask exit/raise nodes back to old state
    mask_nodes_k<<<BATCH * 2, 128, 0, stream>>>(c0n, h0n, c1n, h1n,
                                                hs_c0, hs_h0, hs_c1, hs_h1,
                                                exit_nidx, raise_nidx);

    // 4) decision heads + ip scatter
    decisions_k<<<MROWS / 8, 256, 0, stream>>>(c0n, h0n, c1n, h1n,
                                               W_raise, b_raise, W_branch, b_branch,
                                               ip_old, exit_nidx, raise_nidx,
                                               true_idx, false_idx, raise_idx,
                                               wr, wt, wf, ipn);

    // 5) weighted segment-sum of states into output
    aggregate_k<<<(int)(((size_t)MROWS * HDIM + 255) / 256), 256, 0, stream>>>(
        c0n, h0n, c1n, h1n, wr, wt, wf, true_idx, false_idx, raise_idx, out);

    // 6) finalize: divide by denom, keep_old_if_done, pack ip
    finalize_k<<<(int)(((size_t)MROWS * OUTC + 255) / 256), 256, 0, stream>>>(
        out, ipn, ip_old, hs_c0, hs_h0, hs_c1, hs_h1, step_lim, cur_step);
}
